// CausalFreqMixer_58171037057486
// MI455X (gfx1250) — matrix operand order repaired and checked by reading, not compiled, not run on hardware
//
#include <hip/hip_runtime.h>
#include <cstdint>

typedef __attribute__((ext_vector_type(16))) __bf16 v16bf;
typedef __attribute__((ext_vector_type(8)))  float  v8f;

#define DMODEL 1024
#define SEQ    2048
#define BATCH  4
#define NPROJ  3072
#define MROWS  (BATCH*SEQ)   /* 8192 */

// ---------------- helpers ----------------

__device__ __forceinline__ unsigned short f2bf(float f) {
  uint32_t u = __float_as_uint(f);
  u += 0x7FFFu + ((u >> 16) & 1u);          // round-to-nearest-even
  return (unsigned short)(u >> 16);
}

__device__ __forceinline__ uint32_t lds_off32(const void* p) {
  // generic pointer to LDS: low 32 bits are the LDS byte offset (ISA flat->LDS rule)
  return (uint32_t)(uintptr_t)p;
}

__device__ __forceinline__ void async_ld_b128(uint32_t lds, uint32_t goff,
                                              unsigned long long sbase) {
  // CDNA5 async global -> LDS copy (GVS mode: saddr64 + voffset32), tracked by ASYNCcnt
  asm volatile("global_load_async_to_lds_b128 %0, %1, %2"
               :: "v"(lds), "v"(goff), "s"(sbase) : "memory");
}

__device__ __forceinline__ void wait_async_le4() {
  asm volatile("s_wait_asynccnt 4" ::: "memory");
}
__device__ __forceinline__ void wait_async_0() {
  asm volatile("s_wait_asynccnt 0" ::: "memory");
}

// ---------------- pack / convert kernels ----------------

__global__ void pack_x(const float* __restrict__ x, unsigned short* __restrict__ xb) {
  int i = blockIdx.x * 256 + threadIdx.x;            // 8192*1024 elements
  xb[i] = f2bf(x[i]);
}

__global__ void pack_wt(const float* __restrict__ w, unsigned short* __restrict__ wt) {
  int i = blockIdx.x * 256 + threadIdx.x;            // 3072*1024 elements, out [e][d]
  int e = i >> 10, d = i & 1023;
  wt[i] = f2bf(w[(size_t)d * NPROJ + e]);
}

__global__ void pack_fr(const float* __restrict__ f, unsigned short* __restrict__ fr) {
  int i = blockIdx.x * 256 + threadIdx.x;            // 2*1024*2048, out [s][d][t_rev]
  int s = i >> 21;
  int r = i & ((1 << 21) - 1);
  int d = r >> 11;
  int t = r & 2047;
  fr[i] = f2bf(f[((size_t)s * SEQ + (2047 - t)) * DMODEL + d]);
}

// ---------------- projection GEMM (bf16 WMMA, async-LDS double buffered) ----------------

__global__ __launch_bounds__(256) void proj_gemm(
    const unsigned short* __restrict__ xb,   // [8192][1024] bf16
    const unsigned short* __restrict__ wt,   // [3072][1024] bf16 (W^T)
    const float* __restrict__ bias,          // [3072]
    unsigned short* __restrict__ vbf,        // [4][1024][2048] bf16  (p chunk 0)
    float* __restrict__ g1,                  // [4][1024][2048] f32   (p chunk 1)
    float* __restrict__ g2)                  // [4][1024][2048] f32   (p chunk 2)
{
  __shared__ unsigned short As[2][128 * 32];
  __shared__ unsigned short Bs[2][128 * 32];

  const int m0 = blockIdx.y * 128;
  const int n0 = blockIdx.x * 128;
  const int t  = threadIdx.x;
  const int lane  = t & 31;
  const int wid   = t >> 5;
  const int wm = wid & 3, wn = wid >> 2;
  const int mbase = wm * 32, nbase = wn * 64;
  const int mlane = lane & 15;
  const int hi    = lane >> 4;

  const unsigned long long xs = (unsigned long long)xb;
  const unsigned long long ws = (unsigned long long)wt;
  const int row = t >> 1, half = t & 1;

  v8f acc[2][4];
#pragma unroll
  for (int i = 0; i < 2; ++i)
#pragma unroll
    for (int j = 0; j < 4; ++j)
      acc[i][j] = (v8f){0.f,0.f,0.f,0.f,0.f,0.f,0.f,0.f};

  auto stage_load = [&](int buf, int kk) {
    const int k0 = kk * 32;
    uint32_t aL = lds_off32(&As[buf][row * 32 + half * 16]);
    uint32_t aG = (uint32_t)((((m0 + row) * 1024) + k0 + half * 16) * 2);
    async_ld_b128(aL,      aG,      xs);
    async_ld_b128(aL + 16, aG + 16, xs);
    uint32_t bL = lds_off32(&Bs[buf][row * 32 + half * 16]);
    uint32_t bG = (uint32_t)((((n0 + row) * 1024) + k0 + half * 16) * 2);
    async_ld_b128(bL,      bG,      ws);
    async_ld_b128(bL + 16, bG + 16, ws);
  };

  auto compute = [&](int buf) {
    union Frag { uint4 q[2]; v16bf v; };
    Frag a[2], b[4];
    const char* Ab = (const char*)&As[buf][0];
    const char* Bb = (const char*)&Bs[buf][0];
#pragma unroll
    for (int i = 0; i < 2; ++i) {
      const char* p = Ab + (size_t)(mbase + i * 16 + mlane) * 64 + hi * 16;
      a[i].q[0] = *(const uint4*)p;
      a[i].q[1] = *(const uint4*)(p + 32);
    }
#pragma unroll
    for (int j = 0; j < 4; ++j) {
      const char* p = Bb + (size_t)(nbase + j * 16 + mlane) * 64 + hi * 16;
      b[j].q[0] = *(const uint4*)p;
      b[j].q[1] = *(const uint4*)(p + 32);
    }
#pragma unroll
    for (int i = 0; i < 2; ++i)
#pragma unroll
      for (int j = 0; j < 4; ++j)
        acc[i][j] = __builtin_amdgcn_wmma_f32_16x16x32_bf16(
            false, a[i].v, false, b[j].v, (short)0, acc[i][j], false, false);
  };

  const int nk = 1024 / 32;
  stage_load(0, 0);
  for (int kk = 0; kk < nk; ++kk) {
    const int buf = kk & 1;
    if (kk + 1 < nk) { stage_load(buf ^ 1, kk + 1); wait_async_le4(); }
    else             { wait_async_0(); }
    __syncthreads();
    compute(buf);
    __syncthreads();
  }

  // epilogue: bias add, scatter to chunk-specific layouts
  float biasv[4];
#pragma unroll
  for (int j = 0; j < 4; ++j) biasv[j] = bias[n0 + nbase + j * 16 + mlane];

#pragma unroll
  for (int i = 0; i < 2; ++i) {
    const int gm = m0 + mbase + i * 16 + 8 * hi;   // global row (b,l)
    const int b  = gm >> 11;
    const int l  = gm & 2047;
#pragma unroll
    for (int j = 0; j < 4; ++j) {
      const int e = n0 + nbase + j * 16 + mlane;   // global col
      v8f cc = acc[i][j];
      float o[8];
#pragma unroll
      for (int r = 0; r < 8; ++r) o[r] = cc[r] + biasv[j];
      if (e < 1024) {                               // v0 -> bf16 [b][d][t]
        const size_t base = ((((size_t)b << 10) + e) << 11) + l;
        uint32_t pk[4];
#pragma unroll
        for (int r = 0; r < 4; ++r)
          pk[r] = (uint32_t)f2bf(o[2 * r]) | ((uint32_t)f2bf(o[2 * r + 1]) << 16);
        *(uint4*)&vbf[base] = make_uint4(pk[0], pk[1], pk[2], pk[3]);
      } else {                                      // gates -> f32 [b][d][l]
        float* g = (e < 2048) ? g1 : g2;
        const int dch = e & 1023;
        const size_t base = ((((size_t)b << 10) + dch) << 11) + l;
        *(float4*)&g[base]     = make_float4(o[0], o[1], o[2], o[3]);
        *(float4*)&g[base + 4] = make_float4(o[4], o[5], o[6], o[7]);
      }
    }
  }
}

// ---------------- causal depthwise long-conv stage (block-Toeplitz WMMA) ----------------

__global__ __launch_bounds__(256) void conv_stage(
    const unsigned short* __restrict__ vin,  // [4][1024][2048] bf16
    const unsigned short* __restrict__ fr,   // [1024][2048] bf16 reversed filter (stage slice)
    const float* __restrict__ gate,          // [4][1024][2048] f32
    unsigned short* __restrict__ vout,       // bf16 [b][d][t] (stage 0)
    float* __restrict__ fout,                // f32  [b][l][d] (final stage)
    int is_final)
{
  __shared__ unsigned short fR[2048 + 80];       // reversed filter + zero tail (causal taps)
  __shared__ unsigned short vS[4][2048 + 32];    // whole channel of v, 4 batches

  const int d = blockIdx.x;
  const int t = threadIdx.x;

  { // async global -> LDS staging of filter + v
    uint32_t fl = lds_off32(&fR[t * 8]);
    uint32_t fg = (uint32_t)(((size_t)d * 2048 + t * 8) * 2);
    async_ld_b128(fl, fg, (unsigned long long)fr);
    const int b = t >> 6, w = t & 63;
    uint32_t bl = lds_off32(&vS[b][w * 32]);
    uint32_t bg = (uint32_t)((((((size_t)b << 10) + d) << 11) + w * 32) * 2);
#pragma unroll
    for (int c = 0; c < 4; ++c)
      async_ld_b128(bl + c * 16, bg + c * 16, (unsigned long long)vin);
  }
  if (t < 40) ((uint32_t*)&fR[2048])[t] = 0u;                      // zero pad (f[neg]=0)
  if (t < 64) { int b = t >> 4, j = t & 15; ((uint32_t*)&vS[b][2048])[j] = 0u; }
  wait_async_0();
  __syncthreads();

  const int lane = t & 31;
  const int wid  = t >> 5;
  const int m    = lane & 15;
  const int grp  = lane >> 4;
  const int kbA  = grp * 8;       // A frag K groups: {0..7,16..23} / {8..15,24..31}
  const int kbB  = grp * 8;       // B frag K groups: {0..7,16..23} / {8..15,24..31}

  for (int li = 0; li < 16; ++li) {
    const int lt = wid + 8 * li;          // interleaved l-tiles balance the triangle
    const int l0 = lt * 16;
    v8f c = (v8f){0.f,0.f,0.f,0.f,0.f,0.f,0.f,0.f};
    const int nk = (l0 + 16 + 31) >> 5;
    for (int kk = 0; kk < nk; ++kk) {
      const int t0 = kk * 32;
      // A = Toeplitz(f) tile: A[m][k] = f[l0+m - (t0+k)]; reversed storage makes
      // each packed bf16 pair one 32-bit LDS read.
      union { uint32_t u[8]; v16bf v; } A;
      const int i0 = 2047 - (l0 + m) + t0;
#pragma unroll
      for (int j2 = 0; j2 < 4; ++j2) {
        uint32_t lo, hiu;
        __builtin_memcpy(&lo,  &fR[i0 + kbA + 2 * j2], 4);
        __builtin_memcpy(&hiu, &fR[i0 + kbA + 16 + 2 * j2], 4);
        A.u[j2] = lo; A.u[4 + j2] = hiu;
      }
      // B[k][n] = v[b_n][t0+k]; lanes n>=4 read guaranteed zeros
      union { uint4 q[2]; v16bf v; } B;
      const unsigned short* src = (m < 4) ? &vS[m][t0 + kbB] : &fR[2048];
      B.q[0] = *(const uint4*)src;
      B.q[1] = *(const uint4*)(src + 16);
      c = __builtin_amdgcn_wmma_f32_16x16x32_bf16(
          false, A.v, false, B.v, (short)0, c, false, false);
    }
    // epilogue: gate multiply + store
    if (m < 4) {
      const int b = m;
      const size_t base = ((((size_t)b << 10) + d) << 11) + (size_t)(l0 + 8 * grp);
      const float4 gl0 = *(const float4*)&gate[base];
      const float4 gl1 = *(const float4*)&gate[base + 4];
      float o[8];
      o[0] = c[0] * gl0.x; o[1] = c[1] * gl0.y; o[2] = c[2] * gl0.z; o[3] = c[3] * gl0.w;
      o[4] = c[4] * gl1.x; o[5] = c[5] * gl1.y; o[6] = c[6] * gl1.z; o[7] = c[7] * gl1.w;
      if (is_final) {
        const int lb = l0 + 8 * grp;
#pragma unroll
        for (int r = 0; r < 8; ++r)
          fout[(((size_t)b << 11) + (lb + r)) * 1024 + d] = o[r];
      } else {
        uint32_t pk[4];
#pragma unroll
        for (int r = 0; r < 4; ++r)
          pk[r] = (uint32_t)f2bf(o[2 * r]) | ((uint32_t)f2bf(o[2 * r + 1]) << 16);
        *(uint4*)&vout[base] = make_uint4(pk[0], pk[1], pk[2], pk[3]);
      }
    }
  }
}

// ---------------- host launcher ----------------

extern "C" void kernel_launch(void* const* d_in, const int* in_sizes, int n_in,
                              void* d_out, int out_size, void* d_ws, size_t ws_size,
                              hipStream_t stream) {
  const float* x  = (const float*)d_in[0];   // [4][2048][1024]
  const float* pw = (const float*)d_in[1];   // [1024][3072]
  const float* pb = (const float*)d_in[2];   // [3072]
  const float* ft = (const float*)d_in[3];   // [2][1][2048][1024]
  float* out = (float*)d_out;                // [4][2048][1024]

  char* ws = (char*)d_ws;
  unsigned short* xb = (unsigned short*)(ws + 0);           // 16 MiB
  unsigned short* wt = (unsigned short*)(ws + 16777216);    // 6 MiB
  unsigned short* fr = (unsigned short*)(ws + 23068672);    // 8 MiB (2 stages)
  unsigned short* v0 = (unsigned short*)(ws + 31457280);    // 16 MiB
  unsigned short* v1 = (unsigned short*)(ws + 48234496);    // 16 MiB
  float*          g1 = (float*)(ws + 65011712);             // 32 MiB
  float*          g2 = (float*)(ws + 98566144);             // 32 MiB

  pack_x <<<32768, 256, 0, stream>>>(x, xb);
  pack_wt<<<12288, 256, 0, stream>>>(pw, wt);
  pack_fr<<<16384, 256, 0, stream>>>(ft, fr);

  proj_gemm<<<dim3(NPROJ / 128, MROWS / 128), 256, 0, stream>>>(xb, wt, pb, v0, g1, g2);

  // stage 0: v1 = conv(v0, f0) * g1
  conv_stage<<<1024, 256, 0, stream>>>(v0, fr, g1, v1, out, 0);
  // stage 1: out = conv(v1, f1) * g2
  conv_stage<<<1024, 256, 0, stream>>>(v1, fr + (size_t)1024 * 2048, g2, v1, out, 1);
}
